// TfmrAttention_41489384079592
// MI455X (gfx1250) — compile-verified, matrix operand order checked by reading, not run on hardware
//
#include <hip/hip_runtime.h>

// ---------------- problem constants (from reference) ----------------
#define B_  2
#define S_  2048
#define D_  1024
#define H_  16
#define DH_ 64

typedef __attribute__((ext_vector_type(16))) __bf16 v16bf;
typedef __attribute__((ext_vector_type(8)))  float  v8f;

enum { EPI_BF16 = 0, EPI_F32 = 1, EPI_SPLIT = 2, EPI_SPLIT_T = 3,
       EPI_SCORES = 4, EPI_MERGE = 5 };

struct GP {
  const void*           A;     // bf16 (u16 raw) or fp32, [M,K] row-major (lda)
  const unsigned short* Bm;    // bf16 raw, stored [N,K] row-major (ldb) == B^T
  const float*          bias;  // fp32 [N] or nullptr
  void*                 C;
  long long lda, ldb, ldc;
  long long sAz, sBz, sCz;     // per-blockIdx.z element strides
  int M, N, K;
  int epi;
  float scale;
};

static __device__ __forceinline__ unsigned short f2bf(float f) {
  unsigned int x = __float_as_uint(f);
  x += 0x7FFFu + ((x >> 16) & 1u);            // round-to-nearest-even
  return (unsigned short)(x >> 16);
}

// ---- CDNA5 async copy: global -> LDS, 16B per lane, tracked by ASYNCcnt ----
static __device__ __forceinline__ void async_b128(const void* lds_ptr, const void* gptr) {
  unsigned int loff = (unsigned int)(size_t)lds_ptr;          // low 32 bits = LDS offset
  unsigned long long ga = (unsigned long long)(size_t)gptr;
  asm volatile("global_load_async_to_lds_b128 %0, %1, off"
               :: "v"(loff), "v"(ga) : "memory");
}
template <int N>
static __device__ __forceinline__ void wait_async() {
  if constexpr (N == 0) asm volatile("s_wait_asynccnt 0x0" ::: "memory");
  else if constexpr (N == 1) asm volatile("s_wait_asynccnt 0x1" ::: "memory");
  else if constexpr (N == 2) asm volatile("s_wait_asynccnt 0x2" ::: "memory");
  else if constexpr (N == 3) asm volatile("s_wait_asynccnt 0x3" ::: "memory");
  else asm volatile("s_wait_asynccnt 0x4" ::: "memory");
}

// ---------------- double-buffered async bf16 WMMA GEMM ----------------
#define BM 128
#define BK 32
#define PK 40      // LDS pitch: rows 80B -> 16B-aligned for b128, conflict-free

union FragU { v16bf v; unsigned int u[8]; };

template <int TBN, bool AF32>
__global__ __launch_bounds__(256) void gemm_wmma(GP p) {
  constexpr int WN_WAVES = TBN / 32;           // 4 (TBN=128) or 2 (TBN=64)
  constexpr int WM_WAVES = 8 / WN_WAVES;       // 2 or 4
  constexpr int MT = (BM / WM_WAVES) / 16;     // 4 or 2 m-tiles per wave
  constexpr int G = (AF32 ? 0 : 2) + (TBN == 128 ? 2 : 1);  // async b128/thread

  __shared__ unsigned short As[2][BM * PK];
  __shared__ unsigned short Bs[2][TBN * PK];

  const int tid   = threadIdx.x;
  const int z     = blockIdx.z;
  const int tileM = blockIdx.y * BM;
  const int tileN = blockIdx.x * TBN;

  const unsigned short* Ab = (const unsigned short*)p.A + (size_t)z * p.sAz;
  const float*          Af = (const float*)p.A          + (size_t)z * p.sAz;
  const unsigned short* Bb = p.Bm + (size_t)z * p.sBz;

  const int wave  = tid >> 5, lane = tid & 31;
  const int waveM = wave / WN_WAVES, waveN = wave % WN_WAVES;
  const int lrow  = lane & 15, khalf = lane >> 4;

  // staging maps: A is 128x32 (16 elems/thread); B is TBNx32
  const int ar = tid >> 1, ac = (tid & 1) << 4;

  auto issueA = [&](int buf, int k0) {
    const unsigned short* src = Ab + (size_t)(tileM + ar) * p.lda + k0 + ac;
    async_b128(&As[buf][ar * PK + ac], src);
    async_b128(&As[buf][ar * PK + ac + 8], src + 8);
  };
  auto issueB = [&](int buf, int k0) {
    if constexpr (TBN == 128) {
      const int bn = tid >> 1, bc = (tid & 1) << 4;
      const unsigned short* src = Bb + (size_t)(tileN + bn) * p.ldb + k0 + bc;
      async_b128(&Bs[buf][bn * PK + bc], src);
      async_b128(&Bs[buf][bn * PK + bc + 8], src + 8);
    } else {
      const int bn = tid >> 2, bc = (tid & 3) << 3;
      const unsigned short* src = Bb + (size_t)(tileN + bn) * p.ldb + k0 + bc;
      async_b128(&Bs[buf][bn * PK + bc], src);
    }
  };

  v8f acc[MT][2];
#pragma unroll
  for (int i = 0; i < MT; ++i)
#pragma unroll
    for (int j = 0; j < 2; ++j)
#pragma unroll
      for (int r = 0; r < 8; ++r) acc[i][j][r] = 0.0f;

  const int nk = p.K / BK;
  float4 fa[4];
  if constexpr (AF32) {              // prologue: fp32 A tile 0 -> regs
    const float* src = Af + (size_t)(tileM + ar) * p.lda + ac;
#pragma unroll
    for (int j = 0; j < 4; ++j) fa[j] = *(const float4*)(src + 4 * j);
  } else {
    issueA(0, 0);
  }
  issueB(0, 0);

  for (int kt = 0; kt < nk; ++kt) {
    const int cur = kt & 1, nxt = cur ^ 1;
    if constexpr (AF32) {            // convert + store A tile kt into LDS[cur]
      unsigned short* dst = &As[cur][ar * PK + ac];
#pragma unroll
      for (int j = 0; j < 4; ++j) {
        dst[4 * j + 0] = f2bf(fa[j].x); dst[4 * j + 1] = f2bf(fa[j].y);
        dst[4 * j + 2] = f2bf(fa[j].z); dst[4 * j + 3] = f2bf(fa[j].w);
      }
    }
    if (kt + 1 < nk) {               // issue next tile, keep next's G in flight
      const int k1 = (kt + 1) * BK;
      if constexpr (AF32) {
        const float* src = Af + (size_t)(tileM + ar) * p.lda + k1 + ac;
#pragma unroll
        for (int j = 0; j < 4; ++j) fa[j] = *(const float4*)(src + 4 * j);
      } else {
        issueA(nxt, k1);
      }
      issueB(nxt, k1);
      wait_async<G>();               // current tile's async writes done
    } else {
      wait_async<0>();
    }
    __syncthreads();

    // ---- fragments (CDNA5 16-bit layout) + 8/4 WMMAs ----
    FragU af[MT], bfg[2];
#pragma unroll
    for (int im = 0; im < MT; ++im) {
      const unsigned short* ap =
          &As[cur][(waveM * (MT * 16) + im * 16 + lrow) * PK + khalf * 8];
#pragma unroll
      for (int v = 0; v < 8; ++v)
        af[im].u[v] = *(const unsigned int*)(ap + ((v & 4) << 2) + ((v & 3) << 1));
    }
#pragma unroll
    for (int in = 0; in < 2; ++in) {
      const unsigned short* bp =
          &Bs[cur][(waveN * 32 + in * 16 + lrow) * PK + khalf * 8];
#pragma unroll
      for (int v = 0; v < 8; ++v)
        bfg[in].u[v] = *(const unsigned int*)(bp + ((v & 4) << 2) + ((v & 3) << 1));
    }
#pragma unroll
    for (int im = 0; im < MT; ++im)
#pragma unroll
      for (int in = 0; in < 2; ++in)
        acc[im][in] = __builtin_amdgcn_wmma_f32_16x16x32_bf16(
            false, af[im].v, false, bfg[in].v, (short)0, acc[im][in], false, false);
    __syncthreads();
  }

  // ---- epilogue: C VGPR r -> M = r + 8*khalf, N = lane&15 ----
#pragma unroll
  for (int im = 0; im < MT; ++im)
#pragma unroll
    for (int in = 0; in < 2; ++in)
#pragma unroll
      for (int r = 0; r < 8; ++r) {
        int gm = tileM + waveM * (MT * 16) + im * 16 + khalf * 8 + r;
        int gn = tileN + waveN * 32 + in * 16 + lrow;
        float val = acc[im][in][r];
        if (p.bias) val += p.bias[gn];
        switch (p.epi) {
          case EPI_BF16:
            ((unsigned short*)p.C)[(size_t)gm * p.ldc + gn] = f2bf(val); break;
          case EPI_F32:
            ((float*)p.C)[(size_t)gm * p.ldc + gn] = val; break;
          case EPI_SPLIT: {   // [B*S, D] -> [B,H,S,DH]
            int b = gm >> 11, s = gm & (S_ - 1);
            int h = gn >> 6,  d = gn & (DH_ - 1);
            ((unsigned short*)p.C)[(((size_t)(b * H_ + h)) * S_ + s) * DH_ + d] = f2bf(val);
          } break;
          case EPI_SPLIT_T: { // [B*S, D] -> [B,H,DH,S]  (V^T for PV GEMM)
            int b = gm >> 11, s = gm & (S_ - 1);
            int h = gn >> 6,  d = gn & (DH_ - 1);
            ((unsigned short*)p.C)[(((size_t)(b * H_ + h)) * DH_ + d) * S_ + s] = f2bf(val);
          } break;
          case EPI_SCORES: {  // strictly-causal mask, fp32 out
            float o = (gn < gm) ? val * p.scale : -10000.0f;
            ((float*)p.C)[(size_t)z * p.sCz + (size_t)gm * p.ldc + gn] = o;
          } break;
          case EPI_MERGE: {   // per-(b,h) [S,DH] -> merged [B,S,D]
            int b = z / H_, h = z - b * H_;
            ((unsigned short*)p.C)[((size_t)(b * S_ + gm)) * D_ + h * DH_ + gn] = f2bf(val);
          } break;
        }
      }
}

// ---------------- helpers ----------------
__global__ __launch_bounds__(256) void cvt_k(const float* __restrict__ in,
                                             unsigned short* __restrict__ out, int n) {
  int i = blockIdx.x * 256 + threadIdx.x;
  if (i < n) out[i] = f2bf(in[i]);
}

// fp32 [K,N] -> bf16 [N,K] (LDS-tiled transpose, block 32x8)
__global__ void cvtT_k(const float* __restrict__ in, unsigned short* __restrict__ out,
                       int K, int N) {
  __shared__ unsigned short t[32][33];
  int k0 = blockIdx.x << 5, n0 = blockIdx.y << 5;
  int x = threadIdx.x;
#pragma unroll
  for (int yy = threadIdx.y; yy < 32; yy += 8)
    t[yy][x] = f2bf(in[(size_t)(k0 + yy) * N + n0 + x]);
  __syncthreads();
#pragma unroll
  for (int yy = threadIdx.y; yy < 32; yy += 8)
    out[(size_t)(n0 + yy) * K + k0 + x] = t[x][yy];
}

// row softmax over S_=2048 (in-place fp32 + optional bf16 copy for PV GEMM)
__global__ __launch_bounds__(256) void softmax_k(float* __restrict__ w,
                                                 unsigned short* __restrict__ wb) {
  float* row = w + (size_t)blockIdx.x * S_;
  const int t = threadIdx.x;
  float e[8];
  float mx = -3.4e38f;
#pragma unroll
  for (int i = 0; i < 8; ++i) { e[i] = row[t + i * 256]; mx = fmaxf(mx, e[i]); }
  __shared__ float red[256];
  red[t] = mx; __syncthreads();
  for (int s = 128; s > 0; s >>= 1) { if (t < s) red[t] = fmaxf(red[t], red[t + s]); __syncthreads(); }
  mx = red[0]; __syncthreads();
  float sum = 0.0f;
#pragma unroll
  for (int i = 0; i < 8; ++i) { e[i] = __expf(e[i] - mx); sum += e[i]; }
  red[t] = sum; __syncthreads();
  for (int s = 128; s > 0; s >>= 1) { if (t < s) red[t] += red[t + s]; __syncthreads(); }
  float inv = 1.0f / red[0];
  unsigned short* rowb = wb ? wb + (size_t)blockIdx.x * S_ : nullptr;
#pragma unroll
  for (int i = 0; i < 8; ++i) {
    float r = e[i] * inv;
    row[t + i * 256] = r;
    if (rowb) rowb[t + i * 256] = f2bf(r);
  }
}

// ---------------- launch ----------------
extern "C" void kernel_launch(void* const* d_in, const int* in_sizes, int n_in,
                              void* d_out, int out_size, void* d_ws, size_t ws_size,
                              hipStream_t stream) {
  (void)in_sizes; (void)n_in; (void)out_size;
  const float* hs  = (const float*)d_in[0];
  const float* caw = (const float*)d_in[1];
  const float* cab = (const float*)d_in[2];
  const float* wqw = (const float*)d_in[3];
  const float* wqb = (const float*)d_in[4];
  const float* wkw = (const float*)d_in[5];
  const float* wkb = (const float*)d_in[6];
  const float* wvw = (const float*)d_in[7];
  const float* wvb = (const float*)d_in[8];
  const float* cpw = (const float*)d_in[9];
  const float* cpb = (const float*)d_in[10];

  float* out = (float*)d_out;
  const size_t OUT_W = (size_t)B_ * S_ * D_;
  float* scores = out + OUT_W;                 // attn_weights [B,H,S,S]

  char* ws = (char*)d_ws; size_t off = 0;
  auto alloc16 = [&](size_t elems) -> unsigned short* {
    unsigned short* p = (unsigned short*)(ws + off);
    off = (off + elems * 2 + 255) & ~(size_t)255;
    return p;
  };
  unsigned short* hsb  = alloc16((size_t)B_ * S_ * D_);
  unsigned short* cawt = alloc16((size_t)3 * D_ * D_);
  unsigned short* wqt  = alloc16((size_t)D_ * D_);
  unsigned short* wkt  = alloc16((size_t)D_ * D_);
  unsigned short* wvt  = alloc16((size_t)D_ * D_);
  unsigned short* cpwt = alloc16((size_t)D_ * D_);
  unsigned short* qkvb = alloc16((size_t)B_ * S_ * 3 * D_);
  unsigned short* qb   = alloc16((size_t)B_ * H_ * S_ * DH_);
  unsigned short* kb   = alloc16((size_t)B_ * H_ * S_ * DH_);
  unsigned short* vtb  = alloc16((size_t)B_ * H_ * DH_ * S_);
  unsigned short* mrg  = alloc16((size_t)B_ * S_ * D_);
  // optional bf16 copy of probabilities (removes fp32->bf16 VALU from PV loop)
  const size_t pb_elems = (size_t)B_ * H_ * S_ * S_;
  const bool use_pb = (ws_size >= off + pb_elems * 2 + 256);
  unsigned short* pb = use_pb ? alloc16(pb_elems) : nullptr;

  { int n = B_ * S_ * D_;
    cvt_k<<<dim3((n + 255) / 256), 256, 0, stream>>>(hs, hsb, n); }
  dim3 tb(32, 8);
  cvtT_k<<<dim3(D_ / 32, 3 * D_ / 32), tb, 0, stream>>>(caw, cawt, D_, 3 * D_);
  cvtT_k<<<dim3(D_ / 32, D_ / 32),     tb, 0, stream>>>(wqw, wqt, D_, D_);
  cvtT_k<<<dim3(D_ / 32, D_ / 32),     tb, 0, stream>>>(wkw, wkt, D_, D_);
  cvtT_k<<<dim3(D_ / 32, D_ / 32),     tb, 0, stream>>>(wvw, wvt, D_, D_);
  cvtT_k<<<dim3(D_ / 32, D_ / 32),     tb, 0, stream>>>(cpw, cpwt, D_, D_);

  GP p{};
  // 1) qkv = X @ c_attn_w + b   [4096,1024]x[1024,3072]
  p.A = hsb; p.Bm = cawt; p.bias = cab; p.C = qkvb;
  p.lda = D_; p.ldb = D_; p.ldc = 3 * D_;
  p.M = B_ * S_; p.N = 3 * D_; p.K = D_;
  p.epi = EPI_BF16; p.scale = 1.0f;
  gemm_wmma<128, false><<<dim3(p.N / 128, p.M / BM, 1), 256, 0, stream>>>(p);

  // 2) q/k/v projections (A = qkv column slices, lda=3072), split heads
  p.lda = 3 * D_; p.ldb = D_; p.N = D_; p.K = D_;
  p.A = qkvb;          p.Bm = wqt; p.bias = wqb; p.C = qb;  p.epi = EPI_SPLIT;
  gemm_wmma<128, false><<<dim3(p.N / 128, p.M / BM, 1), 256, 0, stream>>>(p);
  p.A = qkvb + D_;     p.Bm = wkt; p.bias = wkb; p.C = kb;  p.epi = EPI_SPLIT;
  gemm_wmma<128, false><<<dim3(p.N / 128, p.M / BM, 1), 256, 0, stream>>>(p);
  p.A = qkvb + 2 * D_; p.Bm = wvt; p.bias = wvb; p.C = vtb; p.epi = EPI_SPLIT_T;
  gemm_wmma<128, false><<<dim3(p.N / 128, p.M / BM, 1), 256, 0, stream>>>(p);

  // 3) scores = Q @ K^T / 8, strictly-causal mask, fp32 -> d_out weights region
  p = GP{};
  p.A = qb; p.Bm = kb; p.bias = nullptr; p.C = scores;
  p.lda = DH_; p.ldb = DH_; p.ldc = S_;
  p.sAz = (long long)S_ * DH_; p.sBz = (long long)S_ * DH_; p.sCz = (long long)S_ * S_;
  p.M = S_; p.N = S_; p.K = DH_;
  p.epi = EPI_SCORES; p.scale = 0.125f;     // 1/sqrt(64)
  gemm_wmma<128, false><<<dim3(S_ / 128, S_ / BM, B_ * H_), 256, 0, stream>>>(p);

  // 4) row softmax (writes attn_weights output + optional bf16 copy)
  softmax_k<<<dim3(B_ * H_ * S_), 256, 0, stream>>>(scores, pb);

  // 5) attn = P @ V  (B = V^T [DH,S]); merge heads
  p = GP{};
  p.Bm = vtb; p.bias = nullptr; p.C = mrg;
  p.ldb = S_; p.ldc = 0;
  p.sAz = (long long)S_ * S_; p.sBz = (long long)DH_ * S_;
  p.M = S_; p.N = DH_; p.K = S_; p.lda = S_;
  p.epi = EPI_MERGE;
  if (use_pb) {
    p.A = pb;
    gemm_wmma<64, false><<<dim3(1, S_ / BM, B_ * H_), 256, 0, stream>>>(p);
  } else {
    p.A = scores;   // fp32 A, converted while staging
    gemm_wmma<64, true><<<dim3(1, S_ / BM, B_ * H_), 256, 0, stream>>>(p);
  }

  // 6) attn_out = merged @ c_proj_w + b  -> fp32 d_out
  p = GP{};
  p.A = mrg; p.Bm = cpwt; p.bias = cpb; p.C = out;
  p.lda = D_; p.ldb = D_; p.ldc = D_;
  p.M = B_ * S_; p.N = D_; p.K = D_;
  p.epi = EPI_F32;
  gemm_wmma<128, false><<<dim3(p.N / 128, p.M / BM, 1), 256, 0, stream>>>(p);
}
